// WindowAttention3D_59072980189210
// MI455X (gfx1250) — matrix operand order repaired and checked by reading, not compiled, not run on hardware
//
#include <hip/hip_runtime.h>
#include <math.h>

#define B_WIN 2048
#define N_TOK 98
#define NPAD  112
#define C_DIM 128
#define H_NUM 4
#define HD    32
#define L_WIN 512
#define M_REL 507   /* 3*13*13 */

#define XSH 136   /* f16 LDS stride for x tile (272B rows: 16B aligned, conflict-free) */
#define SSF 116   /* f32 LDS stride for logits */
#define PSH 136   /* f16 LDS stride for P and O */

typedef __attribute__((ext_vector_type(16))) _Float16 v16h;
typedef __attribute__((ext_vector_type(8)))  _Float16 h8v;
typedef __attribute__((ext_vector_type(8)))  float    v8f;

__device__ __forceinline__ v8f wmma16(v16h a, v16h b, v8f c) {
  return __builtin_amdgcn_wmma_f32_16x16x32_f16(
      /*neg_a=*/false, a, /*neg_b=*/false, b,
      /*c_mod=*/(short)0, c, /*reuse_a=*/false, /*reuse_b=*/false);
}

// Build a v16h fragment from two contiguous 16-byte chunks.
__device__ __forceinline__ v16h ld16(const _Float16* p0, const _Float16* p1) {
  h8v lo = *(const h8v*)p0;
  h8v hi = *(const h8v*)p1;
  return __builtin_shufflevector(lo, hi, 0,1,2,3,4,5,6,7,8,9,10,11,12,13,14,15);
}

__device__ __forceinline__ v8f vzero8() {
  v8f z = {0.f,0.f,0.f,0.f,0.f,0.f,0.f,0.f};
  return z;
}

// ---------------------------------------------------------------------------
// Kernel 0: one-time f32 -> f16 weight conversion (weights then live in L2).
// ---------------------------------------------------------------------------
__global__ __launch_bounds__(256) void cvt_kernel(
    const float* __restrict__ qkv_w, const float* __restrict__ proj_w,
    _Float16* __restrict__ wq, _Float16* __restrict__ wp)
{
  const int i = blockIdx.x * 256 + threadIdx.x;
  if (i < 3 * C_DIM * C_DIM) wq[i] = (_Float16)qkv_w[i];
  if (i < C_DIM * C_DIM)     wp[i] = (_Float16)proj_w[i];
}

// ---------------------------------------------------------------------------
// Kernel 1: QKV projection, one WG per window.
//   q_ws/k_ws: [B, H, NPAD, HD] f16 (pad rows zero; q pre-scaled)
//   vT_ws:     [B, H, HD, 128]  f16 (transposed; pad cols zero)
// ---------------------------------------------------------------------------
__global__ __launch_bounds__(256) void qkv_kernel(
    const float* __restrict__ x,      // [B, N, C]
    const _Float16* __restrict__ wq,  // [3C, C] f16
    const float* __restrict__ qkv_b,  // [3C]
    _Float16* __restrict__ q_ws,
    _Float16* __restrict__ k_ws,
    _Float16* __restrict__ vT_ws)
{
  __shared__ _Float16 xs[NPAD * XSH];
  const int b     = blockIdx.x;
  const int tid   = threadIdx.x;
  const int lane  = tid & 31;
  const int wave  = tid >> 5;
  const int n16   = lane & 15;
  const int kgb   = (lane >> 4) << 3;   // unused (0 or 8); A-frag K base is kb2
  const int kb2   = (lane >> 4) << 4;   // B-frag K base (0 or 16)
  const int mbase = (lane >> 4) << 3;   // C/D row base (0 or 8)

  // Stage x[b] as f16, zeroed pad rows.
  const float* xb = x + (size_t)b * N_TOK * C_DIM;
  for (int i = tid; i < NPAD * C_DIM; i += 256) {
    const int r = i >> 7, c = i & 127;
    xs[r * XSH + c] = (r < N_TOK) ? (_Float16)xb[r * C_DIM + c] : (_Float16)0.f;
  }
  __syncthreads();

  for (int tj = wave; tj < 24; tj += 8) {
    const int col = tj * 16 + n16;
    // Hoist the 4 B fragments (whole K=128 of this weight column).
    v16h bf[4];
    const _Float16* wr = wq + (size_t)col * C_DIM + kb2;
#pragma unroll
    for (int k = 0; k < 4; ++k) bf[k] = ld16(wr + k * 32, wr + k * 32 + 8);

    const float bias = qkv_b[col];
    const int j3 = tj >> 3;                 // 0=q 1=k 2=v
    const int hh = (tj >> 1) & 3;
    const int d  = ((tj & 1) << 4) + n16;
    const float mscale = (j3 == 0) ? 0.17677669529663687f : 1.0f;

    for (int ti = 0; ti < 7; ++ti) {
      v8f acc = vzero8();
      const _Float16* ar = xs + (ti * 16 + n16) * XSH;
#pragma unroll
      for (int k = 0; k < 4; ++k)
        acc = wmma16(ld16(ar + k * 32 + kb2, ar + k * 32 + kb2 + 8), bf[k], acc);

      const bool edge = (ti == 6);          // rows 96..111 tile
      if (j3 == 2) {
        // transposed store: 8 contiguous f16 per lane
        _Float16* dst = vT_ws + (((size_t)b * H_NUM + hh) * HD + d) * 128 + ti * 16 + mbase;
        h8v o;
#pragma unroll
        for (int i = 0; i < 8; ++i) {
          float v = acc[i] + bias;
          if (edge && (mbase + i) >= 2) v = 0.f;   // rows >= 98
          o[i] = (_Float16)v;
        }
        *(h8v*)dst = o;
        if (edge) {                                 // zero m = 112..127 pad
          h8v z;
#pragma unroll
          for (int i = 0; i < 8; ++i) z[i] = (_Float16)0.f;
          *(h8v*)(dst + 16) = z;
        }
      } else {
        _Float16* dst = ((j3 == 0) ? q_ws : k_ws)
                        + (((size_t)b * H_NUM + hh) * NPAD) * HD + d;
#pragma unroll
        for (int i = 0; i < 8; ++i) {
          const int row = ti * 16 + mbase + i;
          float v = (acc[i] + bias) * mscale;
          if (edge && (mbase + i) >= 2) v = 0.f;   // zero pad rows
          dst[(size_t)row * HD] = (_Float16)v;
        }
      }
    }
  }
}

// ---------------------------------------------------------------------------
// Kernel 2: attention + fused output projection, one WG per window.
// ---------------------------------------------------------------------------
__global__ __launch_bounds__(256) void attn_kernel(
    const _Float16* __restrict__ q_ws,
    const _Float16* __restrict__ k_ws,
    const _Float16* __restrict__ vT_ws,
    const float* __restrict__ mask,   // [L, N, N]
    const float* __restrict__ rel,    // [H, M_REL]
    const _Float16* __restrict__ wp,  // [C, C] f16
    const float* __restrict__ proj_b, // [C]
    float* __restrict__ out)          // [B, N, C]
{
  __shared__ float    Sl[NPAD * SSF];   // f32 logits
  __shared__ _Float16 Pl[NPAD * PSH];   // f16 probabilities (padded, pads stay 0)
  __shared__ _Float16 Ol[NPAD * PSH];   // f16 context, heads concatenated

  const int b     = blockIdx.x;
  const int tid   = threadIdx.x;
  const int lane  = tid & 31;
  const int wave  = tid >> 5;
  const int n16   = lane & 15;
  const int kgb   = (lane >> 4) << 3;
  const int kb2   = (lane >> 4) << 4;
  const int mbase = (lane >> 4) << 3;

  // One-time zero of P (rows>=98 and cols>=98 must be exactly 0 for WMMA pads).
  for (int i = tid; i < NPAD * PSH; i += 256) Pl[i] = (_Float16)0.f;

  const _Float16* qb  = q_ws  + (size_t)b * H_NUM * NPAD * HD;
  const _Float16* kkb = k_ws  + (size_t)b * H_NUM * NPAD * HD;
  const _Float16* vTb = vT_ws + (size_t)b * H_NUM * HD * 128;
  const float* maskb  = mask  + (size_t)(b % L_WIN) * N_TOK * N_TOK;
  __syncthreads();

  for (int h = 0; h < H_NUM; ++h) {
    const _Float16* qh = qb  + (size_t)h * NPAD * HD;
    const _Float16* kh = kkb + (size_t)h * NPAD * HD;

    // ---- S = q @ k^T : one WMMA per tile (K = 32 = head dim) ----
    if (wave < 7) {
      const int ti = wave;
      const _Float16* qr = qh + (ti * 16 + n16) * HD;
      const v16h a = ld16(qr + kb2, qr + kb2 + 8);
      for (int tj = 0; tj < 7; ++tj) {
        const _Float16* kr = kh + (tj * 16 + n16) * HD + kb2;
        v8f acc = wmma16(a, ld16(kr, kr + 8), vzero8());
#pragma unroll
        for (int i = 0; i < 8; ++i)
          Sl[(ti * 16 + mbase + i) * SSF + tj * 16 + n16] = acc[i];
      }
    }
    __syncthreads();

    // ---- + relative-position bias (analytic index) + mask, in f32 ----
    for (int e = tid; e < N_TOK * N_TOK; e += 256) {
      const int r = e / N_TOK, c = e % N_TOK;
      const int t1 = r / 49, r1 = r % 49, h1 = r1 / 7, w1 = r1 % 7;
      const int t2 = c / 49, r2 = c % 49, h2 = r2 / 7, w2 = r2 % 7;
      const int idx = (t1 - t2 + 1) * 169 + (h1 - h2 + 6) * 13 + (w1 - w2 + 6);
      Sl[r * SSF + c] += rel[h * M_REL + idx] + maskb[e];
    }
    __syncthreads();

    // ---- wave-parallel row softmax; write P as f16 ----
    for (int r = wave; r < N_TOK; r += 8) {
      float mx = -1e30f;
      for (int c = lane; c < N_TOK; c += 32) mx = fmaxf(mx, Sl[r * SSF + c]);
#pragma unroll
      for (int off = 16; off; off >>= 1) mx = fmaxf(mx, __shfl_xor(mx, off, 32));
      float sum = 0.f;
      for (int c = lane; c < N_TOK; c += 32) {
        const float ev = __expf(Sl[r * SSF + c] - mx);
        Sl[r * SSF + c] = ev;
        sum += ev;
      }
#pragma unroll
      for (int off = 16; off; off >>= 1) sum += __shfl_xor(sum, off, 32);
      const float inv = 1.0f / sum;
      for (int c = lane; c < N_TOK; c += 32)
        Pl[r * PSH + c] = (_Float16)(Sl[r * SSF + c] * inv);
    }
    __syncthreads();

    // ---- O[:, h*32 : h*32+32] = P @ V   (K = 128, vT rows contiguous) ----
    if (wave < 7) {
      const int ti = wave;
      v16h a[4];
      const _Float16* pr = Pl + (ti * 16 + n16) * PSH;
#pragma unroll
      for (int k = 0; k < 4; ++k) a[k] = ld16(pr + k * 32 + kb2, pr + k * 32 + kb2 + 8);
      for (int tj = 0; tj < 2; ++tj) {
        v8f acc = vzero8();
        const _Float16* vr = vTb + ((size_t)h * HD + tj * 16 + n16) * 128 + kb2;
#pragma unroll
        for (int k = 0; k < 4; ++k)
          acc = wmma16(a[k], ld16(vr + k * 32, vr + k * 32 + 8), acc);
#pragma unroll
        for (int i = 0; i < 8; ++i)
          Ol[(ti * 16 + mbase + i) * PSH + h * HD + tj * 16 + n16] = (_Float16)acc[i];
      }
    }
    __syncthreads();
  }

  // ---- fused projection: out[b] = O @ proj_w^T + proj_b (B-frags hoisted) ----
  {
    const int tj  = wave;                 // 8 column tiles, one per wave
    const int col = tj * 16 + n16;
    v16h bf[4];
    const _Float16* wr = wp + (size_t)col * C_DIM + kb2;
#pragma unroll
    for (int k = 0; k < 4; ++k) bf[k] = ld16(wr + k * 32, wr + k * 32 + 8);
    const float bias = proj_b[col];
    float* dst = out + ((size_t)b * N_TOK) * C_DIM + col;

    for (int ti = 0; ti < 7; ++ti) {
      v8f acc = vzero8();
      const _Float16* ar = Ol + (ti * 16 + n16) * PSH;
#pragma unroll
      for (int k = 0; k < 4; ++k)
        acc = wmma16(ld16(ar + k * 32 + kb2, ar + k * 32 + kb2 + 8), bf[k], acc);

      if (ti < 6) {
#pragma unroll
        for (int i = 0; i < 8; ++i)
          dst[(size_t)(ti * 16 + mbase + i) * C_DIM] = acc[i] + bias;
      } else {
#pragma unroll
        for (int i = 0; i < 8; ++i) {
          const int row = 96 + mbase + i;
          if (row < N_TOK) dst[(size_t)row * C_DIM] = acc[i] + bias;
        }
      }
    }
  }
}

// ---------------------------------------------------------------------------
extern "C" void kernel_launch(void* const* d_in, const int* in_sizes, int n_in,
                              void* d_out, int out_size, void* d_ws, size_t ws_size,
                              hipStream_t stream) {
  (void)in_sizes; (void)n_in; (void)out_size; (void)ws_size;
  const float* x     = (const float*)d_in[0];
  const float* mask  = (const float*)d_in[1];
  const float* qkv_w = (const float*)d_in[2];
  const float* qkv_b = (const float*)d_in[3];
  const float* rel   = (const float*)d_in[4];
  const float* projw = (const float*)d_in[5];
  const float* projb = (const float*)d_in[6];
  float* out = (float*)d_out;

  // Workspace layout (f16), ~185 MB total -> mostly L2-resident:
  _Float16* wqh   = (_Float16*)d_ws;                              // 3C*C
  _Float16* wph   = wqh + 3 * C_DIM * C_DIM;                      // C*C
  _Float16* q_ws  = wph + C_DIM * C_DIM;
  _Float16* k_ws  = q_ws + (size_t)B_WIN * H_NUM * NPAD * HD;
  _Float16* vT_ws = k_ws + (size_t)B_WIN * H_NUM * NPAD * HD;

  cvt_kernel<<<(3 * C_DIM * C_DIM + 255) / 256, 256, 0, stream>>>(qkv_w, projw, wqh, wph);
  qkv_kernel<<<B_WIN, 256, 0, stream>>>(x, wqh, qkv_b, q_ws, k_ws, vT_ws);
  attn_kernel<<<B_WIN, 256, 0, stream>>>(q_ws, k_ws, vT_ws, mask, rel, wph, projb, out);
}
